// MultiHeadAttention_20572893348757
// MI455X (gfx1250) — compile-verified, matrix operand order checked by reading, not run on hardware
//
#include <hip/hip_runtime.h>
#include <hip/hip_bf16.h>

// ---------------------------------------------------------------------------
// MHA forward for MI455X (gfx1250, wave32, WMMA + async-to-LDS).
// f32->f16 cvt (weights + activations); WMMA GEMM projections with
// double-buffered GLOBAL_LOAD_ASYNC_TO_LDS_B128 staging (ASYNCcnt);
// flash-attention with online softmax; WMMA O-projection -> f32 output.
// ---------------------------------------------------------------------------

#define B_SZ   2
#define S_LEN  2048
#define DMODEL 1024
#define NHEAD  16
#define HDIM   64
#define MROWS  (B_SZ * S_LEN)   // 4096

typedef __attribute__((ext_vector_type(16))) _Float16 v16h;
typedef __attribute__((ext_vector_type(8)))  _Float16 v8h;
typedef __attribute__((ext_vector_type(8)))  float    v8f;

union AF { v16h v; v8h h[2]; };

static __device__ __forceinline__ v8f wmma16(const AF& a, const AF& b, v8f c) {
    // D = A(16x32 f16) x B(32x16 f16) + C(16x16 f32)
    return __builtin_amdgcn_wmma_f32_16x16x32_f16(
        false, a.v, false, b.v, (short)0, c, false, false);
}

// CDNA5 async copy: global -> LDS, 16B per lane, tracked by ASYNCcnt.
static __device__ __forceinline__ void async_ld_b128(unsigned lds_off,
                                                     const _Float16* g) {
    asm volatile("global_load_async_to_lds_b128 %0, %1, off"
                 :: "v"(lds_off), "v"(g) : "memory");
}

template <int N>
static __device__ __forceinline__ void wait_asynccnt() {
    asm volatile("s_wait_asynccnt %0" :: "n"(N));
}

// ---------------------------------------------------------------------------
// fp32 -> f16 elementwise conversion
// ---------------------------------------------------------------------------
__global__ __launch_bounds__(256) void cvt_f32_f16(const float* __restrict__ src,
                                                   _Float16* __restrict__ dst, int n) {
    int i = blockIdx.x * 256 + threadIdx.x;
    if (i < n) dst[i] = (_Float16)src[i];
}

// ---------------------------------------------------------------------------
// GEMM: Y = X[M,1024](f16) * W^T (W f16 [1024,1024], row n = out col) + bias
// MODE 0: Y f16 head-major [B,H,S,64]      (Q, K)
// MODE 1: Y f16 transposed [B,H,64,S]      (V)
// MODE 2: Y f32 flat [M,1024]              (O projection)
// Block: 256 threads = 8 waves (2M x 4N); block tile 128x256; wave tile 64x64;
// K step 32; double-buffered async-to-LDS staging.
// ---------------------------------------------------------------------------
#define LP 40   // LDS row pitch in halves (80B = 5*16B, keeps b128 aligned)

template <int MODE>
__global__ __launch_bounds__(256) void gemm_wmma(const _Float16* __restrict__ X,
                                                 const _Float16* __restrict__ W16,
                                                 const float* __restrict__ bias,
                                                 void* __restrict__ Yv) {
    constexpr int K = DMODEL;
    const int m0 = blockIdx.y * 128;
    const int n0 = blockIdx.x * 256;
    const int tid  = threadIdx.x;
    const int lane = tid & 31, wave = tid >> 5;
    const int waveM = wave >> 2, waveN = wave & 3;   // 2 x 4 waves
    const int lr = lane & 15, lh = lane >> 4;
    const int kb = lh * 8;

    __shared__ _Float16 Xs[2][128 * LP];   // 20 KB
    __shared__ _Float16 Ws[2][256 * LP];   // 40 KB

    v8f acc[4][4] = {};

    // issue one K-tile's async copies: X 128x32 halves (2/thread),
    // W 256x32 halves (4/thread) -> 6 async instructions per thread.
    auto issue_tile = [&](int buf, int k0) {
        #pragma unroll
        for (int i = 0; i < 2; ++i) {
            int id = tid + i * 256;
            int row = id >> 2, kc = (id & 3) * 8;
            async_ld_b128((unsigned)(uintptr_t)(&Xs[buf][row * LP + kc]),
                          X + (size_t)(m0 + row) * K + k0 + kc);
        }
        #pragma unroll
        for (int i = 0; i < 4; ++i) {
            int id = tid + i * 256;
            int row = id >> 2, kc = (id & 3) * 8;
            async_ld_b128((unsigned)(uintptr_t)(&Ws[buf][row * LP + kc]),
                          W16 + (size_t)(n0 + row) * K + k0 + kc);
        }
    };

    issue_tile(0, 0);

    constexpr int NT = K / 32;   // 32 K-tiles
    for (int it = 0; it < NT; ++it) {
        const int cur = it & 1;
        if (it + 1 < NT) {
            issue_tile(cur ^ 1, (it + 1) * 32);
            wait_asynccnt<6>();   // in-order retire => current tile complete
        } else {
            wait_asynccnt<0>();
        }
        __syncthreads();          // current buffer visible to all waves

        AF a[4], b[4];
        #pragma unroll
        for (int mi = 0; mi < 4; ++mi) {
            int row = waveM * 64 + mi * 16 + lr;
            a[mi].h[0] = *(const v8h*)(&Xs[cur][row * LP + kb]);
            a[mi].h[1] = *(const v8h*)(&Xs[cur][row * LP + kb + 16]);
        }
        #pragma unroll
        for (int ni = 0; ni < 4; ++ni) {
            int row = waveN * 64 + ni * 16 + lr;
            b[ni].h[0] = *(const v8h*)(&Ws[cur][row * LP + kb]);
            b[ni].h[1] = *(const v8h*)(&Ws[cur][row * LP + kb + 16]);
        }
        #pragma unroll
        for (int mi = 0; mi < 4; ++mi)
            #pragma unroll
            for (int ni = 0; ni < 4; ++ni)
                acc[mi][ni] = wmma16(a[mi], b[ni], acc[mi][ni]);

        __syncthreads();          // done reading before next prefetch reuses buf
    }

    // ---- writeback: lane holds (n = lr, m = e + 8*lh) per 16x16 tile ----
    #pragma unroll
    for (int ni = 0; ni < 4; ++ni) {
        int n = n0 + waveN * 64 + ni * 16 + lr;
        float bv = bias[n];
        #pragma unroll
        for (int mi = 0; mi < 4; ++mi) {
            #pragma unroll
            for (int e = 0; e < 8; ++e) {
                int m = m0 + waveM * 64 + mi * 16 + e + 8 * lh;
                float val = acc[mi][ni][e] + bv;
                if (MODE == 2) {
                    ((float*)Yv)[(size_t)m * DMODEL + n] = val;
                } else {
                    int bb = m >> 11, s = m & (S_LEN - 1);
                    int h = n >> 6, d = n & (HDIM - 1);
                    if (MODE == 0) {
                        // [B,H,S,64]
                        ((_Float16*)Yv)[(((size_t)bb * NHEAD + h) * S_LEN + s) * HDIM + d] =
                            (_Float16)val;
                    } else {
                        // [B,H,64,S]  (transposed V for PV B-fragments)
                        ((_Float16*)Yv)[(((size_t)bb * NHEAD + h) * HDIM + d) * S_LEN + s] =
                            (_Float16)val;
                    }
                }
            }
        }
    }
}

// ---------------------------------------------------------------------------
// Flash attention (causal). Grid: (S/128, B*H). 256 threads = 8 waves,
// each wave owns 16 query rows and streams keys in chunks of 32.
// ---------------------------------------------------------------------------
__global__ __launch_bounds__(256) void attn_flash(const _Float16* __restrict__ Q16,
                                                  const _Float16* __restrict__ K16,
                                                  const _Float16* __restrict__ Vt16,
                                                  _Float16* __restrict__ attn16) {
    const int bh   = blockIdx.y;                       // b*NHEAD + h
    const int lane = threadIdx.x & 31, wave = threadIdx.x >> 5;
    const int lr = lane & 15, lh = lane >> 4;
    const int kb = lh * 8;
    const int q0 = blockIdx.x * 128 + wave * 16;

    const _Float16* Qh = Q16 + (size_t)bh * S_LEN * HDIM;
    const _Float16* Kh = K16 + (size_t)bh * S_LEN * HDIM;
    const _Float16* Vh = Vt16 + (size_t)bh * HDIM * S_LEN;   // [64][S]

    __shared__ _Float16 pbuf[8][16][32];   // per-wave P staging (C->A layout)

    // Q A-fragments, kept in registers for the whole sweep
    AF qa[2];
    {
        int qrow = q0 + lr;
        qa[0].h[0] = *(const v8h*)(Qh + (size_t)qrow * HDIM + kb);
        qa[0].h[1] = *(const v8h*)(Qh + (size_t)qrow * HDIM + kb + 16);
        qa[1].h[0] = *(const v8h*)(Qh + (size_t)qrow * HDIM + 32 + kb);
        qa[1].h[1] = *(const v8h*)(Qh + (size_t)qrow * HDIM + 32 + kb + 16);
    }

    float mrow[8], lsum[8];
    #pragma unroll
    for (int e = 0; e < 8; ++e) { mrow[e] = -3.0e38f; lsum[e] = 0.0f; }
    v8f o[4] = {};
    const float scale = 0.125f;   // 1/sqrt(64)

    const int jmax = q0 + 16;     // keys needed: [0, q0+15]
    for (int j0 = 0; j0 < jmax; j0 += 32) {
        // ---- scores: 16 q-rows x 32 keys via 4 WMMAs ----
        AF kf[2][2];
        #pragma unroll
        for (int t = 0; t < 2; ++t)
            #pragma unroll
            for (int c = 0; c < 2; ++c) {
                int krow = j0 + t * 16 + lr;
                kf[t][c].h[0] = *(const v8h*)(Kh + (size_t)krow * HDIM + c * 32 + kb);
                kf[t][c].h[1] = *(const v8h*)(Kh + (size_t)krow * HDIM + c * 32 + kb + 16);
            }
        v8f sc0 = {}, sc1 = {};
        sc0 = wmma16(qa[0], kf[0][0], sc0);
        sc0 = wmma16(qa[1], kf[0][1], sc0);
        sc1 = wmma16(qa[0], kf[1][0], sc1);
        sc1 = wmma16(qa[1], kf[1][1], sc1);

        // ---- scale + causal mask ----
        float sv0[8], sv1[8];
        #pragma unroll
        for (int e = 0; e < 8; ++e) {
            int row = q0 + e + 8 * lh;
            int c0 = j0 + lr, c1 = j0 + 16 + lr;
            sv0[e] = (c0 <= row) ? sc0[e] * scale : -3.0e38f;
            sv1[e] = (c1 <= row) ? sc1[e] * scale : -3.0e38f;
        }

        // ---- online softmax (row reductions across 16-lane halves) ----
        #pragma unroll
        for (int e = 0; e < 8; ++e) {
            float rm = fmaxf(sv0[e], sv1[e]);
            #pragma unroll
            for (int off = 8; off >= 1; off >>= 1)
                rm = fmaxf(rm, __shfl_xor(rm, off, 32));
            float mnew  = fmaxf(mrow[e], rm);
            float alpha = __expf(mrow[e] - mnew);
            float p0 = __expf(sv0[e] - mnew);
            float p1 = __expf(sv1[e] - mnew);
            float rs = p0 + p1;
            #pragma unroll
            for (int off = 8; off >= 1; off >>= 1)
                rs += __shfl_xor(rs, off, 32);
            lsum[e] = lsum[e] * alpha + rs;
            mrow[e] = mnew;
            #pragma unroll
            for (int c = 0; c < 4; ++c) o[c][e] *= alpha;
            // P to LDS in row-major [m][key] for A-fragment reload
            pbuf[wave][e + 8 * lh][lr]      = (_Float16)p0;
            pbuf[wave][e + 8 * lh][16 + lr] = (_Float16)p1;
        }

        // ---- PV: O[16x64] += P[16x32] x V[32x64] ----
        AF pa;
        pa.h[0] = *(const v8h*)&pbuf[wave][lr][kb];
        pa.h[1] = *(const v8h*)&pbuf[wave][lr][kb + 16];
        #pragma unroll
        for (int c = 0; c < 4; ++c) {
            AF vb;
            int dcol = c * 16 + lr;
            vb.h[0] = *(const v8h*)(Vh + (size_t)dcol * S_LEN + j0 + kb);
            vb.h[1] = *(const v8h*)(Vh + (size_t)dcol * S_LEN + j0 + kb + 16);
            o[c] = wmma16(pa, vb, o[c]);
        }
    }

    // ---- finalize: O /= l, write f16 [B,S,1024] ----
    const int bb = bh >> 4, h = bh & (NHEAD - 1);
    float inv[8];
    #pragma unroll
    for (int e = 0; e < 8; ++e) inv[e] = 1.0f / lsum[e];
    #pragma unroll
    for (int c = 0; c < 4; ++c) {
        int col = h * HDIM + c * 16 + lr;
        #pragma unroll
        for (int e = 0; e < 8; ++e) {
            int m = q0 + e + 8 * lh;
            attn16[((size_t)bb * S_LEN + m) * DMODEL + col] = (_Float16)(o[c][e] * inv[e]);
        }
    }
}

// ---------------------------------------------------------------------------
// Host-side orchestration
// ---------------------------------------------------------------------------
extern "C" void kernel_launch(void* const* d_in, const int* in_sizes, int n_in,
                              void* d_out, int out_size, void* d_ws, size_t ws_size,
                              hipStream_t stream) {
    const float* x_q  = (const float*)d_in[0];
    const float* x_k  = (const float*)d_in[1];
    const float* x_v  = (const float*)d_in[2];
    const float* wq_w = (const float*)d_in[3];
    const float* wq_b = (const float*)d_in[4];
    const float* wk_w = (const float*)d_in[5];
    const float* wk_b = (const float*)d_in[6];
    const float* wv_w = (const float*)d_in[7];
    const float* wv_b = (const float*)d_in[8];
    const float* wo_w = (const float*)d_in[9];
    const float* wo_b = (const float*)d_in[10];

    // Workspace layout (f16 halves). Total = 64 MB.
    _Float16* Wq16   = (_Float16*)d_ws;
    _Float16* Wk16   = Wq16 + 1048576;
    _Float16* Wv16   = Wk16 + 1048576;
    _Float16* Wo16   = Wv16 + 1048576;
    _Float16* Xq16   = Wo16 + 1048576;          // [M,1024] f16 activations
    _Float16* Xk16   = Xq16 + 4194304;
    _Float16* Xv16   = Xk16 + 4194304;
    _Float16* Q16    = Xv16 + 4194304;          // [B,H,S,64]
    _Float16* K16    = Q16  + 4194304;          // [B,H,S,64]
    _Float16* Vt16   = K16  + 4194304;          // [B,H,64,S]
    _Float16* attn16 = Vt16 + 4194304;          // [B,S,1024]

    const int NW = DMODEL * DMODEL;             // 1048576
    const int NA = MROWS * DMODEL;              // 4194304
    cvt_f32_f16<<<NW / 256, 256, 0, stream>>>(wq_w, Wq16, NW);
    cvt_f32_f16<<<NW / 256, 256, 0, stream>>>(wk_w, Wk16, NW);
    cvt_f32_f16<<<NW / 256, 256, 0, stream>>>(wv_w, Wv16, NW);
    cvt_f32_f16<<<NW / 256, 256, 0, stream>>>(wo_w, Wo16, NW);
    cvt_f32_f16<<<NA / 256, 256, 0, stream>>>(x_q, Xq16, NA);
    cvt_f32_f16<<<NA / 256, 256, 0, stream>>>(x_k, Xk16, NA);
    cvt_f32_f16<<<NA / 256, 256, 0, stream>>>(x_v, Xv16, NA);

    dim3 gg(DMODEL / 256, MROWS / 128);         // (4, 32)
    gemm_wmma<0><<<gg, 256, 0, stream>>>(Xq16, Wq16, wq_b, Q16);
    gemm_wmma<0><<<gg, 256, 0, stream>>>(Xk16, Wk16, wk_b, K16);
    gemm_wmma<1><<<gg, 256, 0, stream>>>(Xv16, Wv16, wv_b, Vt16);

    attn_flash<<<dim3(S_LEN / 128, B_SZ * NHEAD), 256, 0, stream>>>(Q16, K16, Vt16, attn16);

    gemm_wmma<2><<<gg, 256, 0, stream>>>(attn16, Wo16, wo_b, d_out);
}
